// CrossViewAttention_7662221656162
// MI455X (gfx1250) — matrix operand order repaired and checked by reading, not compiled, not run on hardware
//
#include <hip/hip_runtime.h>
#include <math.h>

typedef __attribute__((ext_vector_type(16))) _Float16 v16h;
typedef __attribute__((ext_vector_type(8)))  float    v8f;
typedef __attribute__((ext_vector_type(4)))  float    v4f;

#define DMODEL 128
#define DHEAD  32
#define NHEADS 4
#define QTOK   2500
#define QPAD   2512        // 157 * 16
#define NKEY   6336        // 396 * 16
#define QTILES 157
#define KTILES 396
#define LN_EPS 1e-5f
#define NEG_INF (-3.0e38f)

union AF { v16h h; v4f f[2]; _Float16 e[16]; };
union CF { v8f  v; float e[8]; };

__device__ __forceinline__ v8f wmma_f16(v16h a, v16h b, v8f c) {
  return __builtin_amdgcn_wmma_f32_16x16x32_f16(false, a, false, b, (short)0, c,
                                                false, false);
}

__device__ __forceinline__ void lds_wait() {
  asm volatile("s_wait_dscnt 0" ::: "memory");
}

// Build four 16x32 f16 A-fragments (K = 0..127) from a row-major fp32 LDS tile.
// A layout (16-bit, 16x32): lane L holds row M = L%16; halves 0..7 -> K = base..base+7,
// halves 8..15 -> K = 16+base..16+base+7, base = 8*(L/16); repeated per 32-K window.
__device__ __forceinline__ void build_afrags(AF* afr, const float* buf, int row, int hi) {
#pragma unroll
  for (int kk = 0; kk < 4; ++kk)
#pragma unroll
    for (int c = 0; c < 2; ++c)
#pragma unroll
      for (int j = 0; j < 8; ++j) {
        int k = kk * 32 + c * 16 + hi * 8 + j;
        afr[kk].e[c * 8 + j] = (_Float16)buf[row * 128 + k];
      }
}

__device__ __forceinline__ void row_stats_128(const float* buf, int lane,
                                              float* smean, float* srsig) {
  if (lane < 16) {
    float s = 0.f, s2 = 0.f;
    for (int d = 0; d < 128; ++d) {
      float v = buf[lane * 128 + d];
      s += v; s2 += v * v;
    }
    float mn  = s * (1.f / 128.f);
    float var = s2 * (1.f / 128.f) - mn * mn;
    smean[lane] = mn;
    srsig[lane] = rsqrtf(var + LN_EPS);
  }
}

// ---------------------------------------------------------------------------
// Kernel 0: fp32 -> f16 weight conversion
// ---------------------------------------------------------------------------
__global__ void cvt_f16_kernel(const float* __restrict__ s, _Float16* __restrict__ d, int n) {
  int i = blockIdx.x * blockDim.x + threadIdx.x;
  if (i < n) d[i] = (_Float16)s[i];
}

// ---------------------------------------------------------------------------
// Kernel 1: LayerNorm + projection (x @ W^T + b) for Q / K / V token streams.
// One wave handles a 16-token tile.  mode 0: q gather [D,2500]; mode 1: k/v
// gather [N,D,h,w].  Output row-major f16 (Q,K) or transposed [feat][tok] (V).
// ---------------------------------------------------------------------------
__global__ __launch_bounds__(32) void ln_proj_kernel(
    const float* __restrict__ src, int mode,
    const float* __restrict__ g, const float* __restrict__ be,
    const _Float16* __restrict__ Wh, const float* __restrict__ bias,
    _Float16* __restrict__ dstR, _Float16* __restrict__ dstT) {
  __shared__ __align__(16) float xb[16 * 128];
  __shared__ float sg[128], sb[128], sbias[128];
  __shared__ float smean[16], srsig[16];

  const int lane = threadIdx.x;
  const int row  = lane & 15, hi = lane >> 4;
  const int t0   = blockIdx.x * 16;

  for (int i = lane; i < 128; i += 32) { sg[i] = g[i]; sb[i] = be[i]; sbias[i] = bias[i]; }

  // Stage 16 tokens x 128 features into LDS (fp32)
  for (int idx = lane; idx < 2048; idx += 32) {
    int t = idx & 15, d = idx >> 4;
    int tok = t0 + t;
    float v = 0.f;
    if (mode == 0) {
      if (tok < QTOK) v = src[d * QTOK + tok];
    } else {
      int n = tok / 1056, rem = tok - n * 1056;      // tiles never cross n (1056 = 66*16)
      v = src[(n * 128 + d) * 1056 + rem];
    }
    xb[t * 128 + d] = v;
  }
  lds_wait();

  row_stats_128(xb, lane, smean, srsig);
  lds_wait();

  const float mn = smean[row], rs = srsig[row];
  AF afr[4];
#pragma unroll
  for (int kk = 0; kk < 4; ++kk)
#pragma unroll
    for (int c = 0; c < 2; ++c)
#pragma unroll
      for (int j = 0; j < 8; ++j) {
        int k = kk * 32 + c * 16 + hi * 8 + j;
        float v = (xb[row * 128 + k] - mn) * rs * sg[k] + sb[k];
        afr[kk].e[c * 8 + j] = (_Float16)v;
      }

  for (int nt = 0; nt < 8; ++nt) {
    CF acc = {};
    const _Float16* wrow = Wh + (nt * 16 + row) * 128 + hi * 8;    // B: N=row, K block
#pragma unroll
    for (int kk = 0; kk < 4; ++kk) {
      AF bf;
      const v4f* wp = (const v4f*)(wrow + kk * 32);
      bf.f[0] = wp[0]; bf.f[1] = wp[2];
      acc.v = wmma_f16(afr[kk].h, bf.h, acc.v);
    }
    const int feat = nt * 16 + row;
    const float bvl = sbias[feat];
#pragma unroll
    for (int r = 0; r < 8; ++r) {
      int tok = t0 + r + 8 * hi;                 // C layout: M = r + 8*(lane/16)
      _Float16 hv = (_Float16)(acc.e[r] + bvl);
      if (dstT) dstT[feat * NKEY + tok] = hv;    // V: [feat][key]
      else      dstR[tok * 128 + feat] = hv;     // Q/K: [tok][feat]
    }
  }
}

// ---------------------------------------------------------------------------
// Kernel 2: flash attention, one wave per (16-query tile, head).
// ---------------------------------------------------------------------------
__global__ __launch_bounds__(32) void attn_kernel(
    const _Float16* __restrict__ Qh, const _Float16* __restrict__ Kh,
    const _Float16* __restrict__ Vt, const float* __restrict__ Wl,
    const int* __restrict__ vis, float* __restrict__ out) {
  __shared__ __align__(16) _Float16 pbuf[16 * 32];

  const int lane = threadIdx.x;
  const int row  = lane & 15, hi = lane >> 4;
  const int qt   = blockIdx.x, head = blockIdx.y;
  const float scale = 0.1767766952966369f;   // 1/sqrt(32)

  AF qf;
  {
    const _Float16* qp = Qh + (qt * 16 + row) * 128 + head * 32 + hi * 8;
    qf.f[0] = ((const v4f*)qp)[0];
    qf.f[1] = ((const v4f*)(qp + 16))[0];
  }

  CF acc0 = {}, acc1 = {};
  float m[8], l[8];
  int qr[8];
#pragma unroll
  for (int r = 0; r < 8; ++r) {
    m[r] = NEG_INF; l[r] = 0.f;
    int q = qt * 16 + r + 8 * hi;
    qr[r] = q < QTOK ? q : QTOK - 1;             // clamp pad rows for W/vis loads
  }

  for (int k0 = 0; k0 < NKEY; k0 += 32) {
    if (k0 + 64 <= NKEY) {                        // next W/vis tiles into cache
      __builtin_prefetch(Wl  + qr[0] * NKEY + k0 + 32, 0, 1);
      __builtin_prefetch(vis + qr[0] * NKEY + k0 + 32, 0, 1);
    }
    AF kf0, kf1;
    {
      const _Float16* p0 = Kh + (k0 + row) * 128 + head * 32 + hi * 8;
      kf0.f[0] = ((const v4f*)p0)[0]; kf0.f[1] = ((const v4f*)p0)[2];
      const _Float16* p1 = Kh + (k0 + 16 + row) * 128 + head * 32 + hi * 8;
      kf1.f[0] = ((const v4f*)p1)[0]; kf1.f[1] = ((const v4f*)p1)[2];
    }
    CF s0 = {}, s1 = {};
    s0.v = wmma_f16(qf.h, kf0.h, s0.v);
    s1.v = wmma_f16(qf.h, kf1.h, s1.v);

    const int kc0 = k0 + row, kc1 = k0 + 16 + row;
#pragma unroll
    for (int r = 0; r < 8; ++r) {
      int base = qr[r] * NKEY;
      float w0 = Wl[base + kc0], w1 = Wl[base + kc1];
      int   v0 = vis[base + kc0], v1 = vis[base + kc1];
      float a = s0.e[r] * scale; a = v0 ? a : NEG_INF; s0.e[r] = a * w0;
      float b = s1.e[r] * scale; b = v1 ? b : NEG_INF; s1.e[r] = b * w1;
    }
#pragma unroll
    for (int r = 0; r < 8; ++r) {
      float rm = fmaxf(s0.e[r], s1.e[r]);
      rm = fmaxf(rm, __shfl_xor(rm, 1, 32));
      rm = fmaxf(rm, __shfl_xor(rm, 2, 32));
      rm = fmaxf(rm, __shfl_xor(rm, 4, 32));
      rm = fmaxf(rm, __shfl_xor(rm, 8, 32));
      float mnew  = fmaxf(m[r], rm);
      float alpha = __expf(m[r] - mnew);
      m[r] = mnew;
      float p0 = __expf(s0.e[r] - mnew);
      float p1 = __expf(s1.e[r] - mnew);
      float rs = p0 + p1;
      rs += __shfl_xor(rs, 1, 32);
      rs += __shfl_xor(rs, 2, 32);
      rs += __shfl_xor(rs, 4, 32);
      rs += __shfl_xor(rs, 8, 32);
      l[r] = l[r] * alpha + rs;
      acc0.e[r] *= alpha; acc1.e[r] *= alpha;
      int M = r + 8 * hi;                        // C layout -> LDS row-major P
      pbuf[M * 32 + row]      = (_Float16)p0;
      pbuf[M * 32 + 16 + row] = (_Float16)p1;
    }
    lds_wait();
    AF pf;                                        // re-read in A layout
    pf.f[0] = ((const v4f*)(pbuf + row * 32 + hi * 8))[0];
    pf.f[1] = ((const v4f*)(pbuf + row * 32 + 16 + hi * 8))[0];

    AF vf0, vf1;                                  // B: N=dh, K=32 keys (Vt contiguous)
    {
      const _Float16* p0 = Vt + (head * 32 + row) * NKEY + k0 + hi * 8;
      vf0.f[0] = ((const v4f*)p0)[0]; vf0.f[1] = ((const v4f*)p0)[2];
      const _Float16* p1 = Vt + (head * 32 + 16 + row) * NKEY + k0 + hi * 8;
      vf1.f[0] = ((const v4f*)p1)[0]; vf1.f[1] = ((const v4f*)p1)[2];
    }
    acc0.v = wmma_f16(pf.h, vf0.h, acc0.v);
    acc1.v = wmma_f16(pf.h, vf1.h, acc1.v);
  }

#pragma unroll
  for (int r = 0; r < 8; ++r) {
    float inv = l[r] > 0.f ? 1.f / l[r] : 0.f;
    int q = qt * 16 + r + 8 * hi;                 // padded buffer, q < 2512
    out[q * 128 + head * 32 + row]      = acc0.e[r] * inv;
    out[q * 128 + head * 32 + 16 + row] = acc1.e[r] * inv;
  }
}

// ---------------------------------------------------------------------------
// Kernel 3: proj + skip + preLN + MLP(GELU) + residual + postLN, per 16 tokens
// ---------------------------------------------------------------------------
__global__ __launch_bounds__(32) void post_kernel(
    const float* __restrict__ ao, const float* __restrict__ skip,
    const _Float16* __restrict__ Wp, const float* __restrict__ bp,
    const float* __restrict__ pre_g, const float* __restrict__ pre_b,
    const _Float16* __restrict__ W1, const float* __restrict__ b1,
    const _Float16* __restrict__ W2, const float* __restrict__ b2,
    const float* __restrict__ post_g, const float* __restrict__ post_b,
    float* __restrict__ out) {
  __shared__ __align__(16) float zb[16 * 128];
  __shared__ __align__(16) float znb[16 * 128];
  __shared__ __align__(16) _Float16 hb[16 * 256];
  __shared__ float smean[16], srsig[16];

  const int lane = threadIdx.x;
  const int row  = lane & 15, hi = lane >> 4;
  const int t0   = blockIdx.x * 16;

  for (int idx = lane; idx < 2048; idx += 32) {   // stage attention output tile
    int d = idx & 127, t = idx >> 7;
    zb[t * 128 + d] = ao[(t0 + t) * 128 + d];
  }
  lds_wait();

  AF afr[4];
  build_afrags(afr, zb, row, hi);                 // proj A-fragments (no LN)

  for (int nt = 0; nt < 8; ++nt) {                // z = ao @ Wp^T + bp + skip
    CF acc = {};
    const _Float16* wrow = Wp + (nt * 16 + row) * 128 + hi * 8;
#pragma unroll
    for (int kk = 0; kk < 4; ++kk) {
      AF bf; const v4f* wp = (const v4f*)(wrow + kk * 32);
      bf.f[0] = wp[0]; bf.f[1] = wp[2];
      acc.v = wmma_f16(afr[kk].h, bf.h, acc.v);
    }
    const int d = nt * 16 + row;
    const float bvl = bp[d];
#pragma unroll
    for (int r = 0; r < 8; ++r) {
      int M = r + 8 * hi;
      int tok = t0 + M; int tokc = tok < QTOK ? tok : QTOK - 1;
      zb[M * 128 + d] = acc.e[r] + bvl + skip[d * QTOK + tokc];
    }
  }
  lds_wait();
  row_stats_128(zb, lane, smean, srsig);          // pre-LN
  lds_wait();
  for (int idx = lane; idx < 2048; idx += 32) {
    int d = idx & 127, t = idx >> 7;
    znb[t * 128 + d] = (zb[t * 128 + d] - smean[t]) * srsig[t] * pre_g[d] + pre_b[d];
  }
  lds_wait();

  AF zfr[4];
  build_afrags(zfr, znb, row, hi);

  for (int nt = 0; nt < 16; ++nt) {               // hmid = gelu(zn @ W1^T + b1)
    CF acc = {};
    const _Float16* wrow = W1 + (nt * 16 + row) * 128 + hi * 8;
#pragma unroll
    for (int kk = 0; kk < 4; ++kk) {
      AF bf; const v4f* wp = (const v4f*)(wrow + kk * 32);
      bf.f[0] = wp[0]; bf.f[1] = wp[2];
      acc.v = wmma_f16(zfr[kk].h, bf.h, acc.v);
    }
    const int n = nt * 16 + row;
    const float bvl = b1[n];
#pragma unroll
    for (int r = 0; r < 8; ++r) {
      float x = acc.e[r] + bvl;
      float ge = 0.5f * x * (1.f + erff(x * 0.7071067811865475f));  // exact gelu
      hb[(r + 8 * hi) * 256 + n] = (_Float16)ge;
    }
  }
  lds_wait();

  AF hfr[8];                                      // A-frags over K = 256
#pragma unroll
  for (int kk = 0; kk < 8; ++kk) {
    hfr[kk].f[0] = ((const v4f*)(hb + row * 256 + kk * 32 + hi * 8))[0];
    hfr[kk].f[1] = ((const v4f*)(hb + row * 256 + kk * 32 + 16 + hi * 8))[0];
  }

  for (int nt = 0; nt < 8; ++nt) {                // z2 = zn + hmid @ W2^T + b2
    CF acc = {};
    const _Float16* wrow = W2 + (nt * 16 + row) * 256 + hi * 8;
#pragma unroll
    for (int kk = 0; kk < 8; ++kk) {
      AF bf; const v4f* wp = (const v4f*)(wrow + kk * 32);
      bf.f[0] = wp[0]; bf.f[1] = wp[2];
      acc.v = wmma_f16(hfr[kk].h, bf.h, acc.v);
    }
    const int d = nt * 16 + row;
    const float bvl = b2[d];
#pragma unroll
    for (int r = 0; r < 8; ++r) {
      int M = r + 8 * hi;
      zb[M * 128 + d] = acc.e[r] + bvl + znb[M * 128 + d];
    }
  }
  lds_wait();
  row_stats_128(zb, lane, smean, srsig);          // post-LN
  lds_wait();
  for (int idx = lane; idx < 2048; idx += 32) {   // transposed store [D][2500]
    int t = idx & 15, d = idx >> 4;
    int tok = t0 + t;
    if (tok < QTOK) {
      float v = (zb[t * 128 + d] - smean[t]) * srsig[t] * post_g[d] + post_b[d];
      out[d * QTOK + tok] = v;
    }
  }
}

// ---------------------------------------------------------------------------
extern "C" void kernel_launch(void* const* d_in, const int* in_sizes, int n_in,
                              void* d_out, int out_size, void* d_ws, size_t ws_size,
                              hipStream_t stream) {
  (void)in_sizes; (void)n_in; (void)out_size; (void)ws_size;
  const float* q      = (const float*)d_in[0];
  const float* k      = (const float*)d_in[1];
  const float* v      = (const float*)d_in[2];
  const float* Wl     = (const float*)d_in[3];
  const int*   vis    = (const int*)  d_in[4];
  const float* skip   = (const float*)d_in[5];
  const float* qn_g   = (const float*)d_in[6],  *qn_b = (const float*)d_in[7];
  const float* kn_g   = (const float*)d_in[8],  *kn_b = (const float*)d_in[9];
  const float* vn_g   = (const float*)d_in[10], *vn_b = (const float*)d_in[11];
  const float* wq     = (const float*)d_in[12], *bq   = (const float*)d_in[13];
  const float* wk     = (const float*)d_in[14], *bk   = (const float*)d_in[15];
  const float* wv     = (const float*)d_in[16], *bv   = (const float*)d_in[17];
  const float* wproj  = (const float*)d_in[18], *bproj= (const float*)d_in[19];
  const float* pre_g  = (const float*)d_in[20], *pre_b= (const float*)d_in[21];
  const float* w1     = (const float*)d_in[22], *b1   = (const float*)d_in[23];
  const float* w2     = (const float*)d_in[24], *b2   = (const float*)d_in[25];
  const float* post_g = (const float*)d_in[26], *post_b=(const float*)d_in[27];

  // Workspace layout (f16-element offsets; everything 16B-aligned)
  _Float16* wqh = (_Float16*)d_ws;
  _Float16* wkh = wqh + 16384;
  _Float16* wvh = wkh + 16384;
  _Float16* wph = wvh + 16384;
  _Float16* w1h = wph + 16384;
  _Float16* w2h = w1h + 32768;
  _Float16* Qh  = w2h + 32768;            // [2512][128]
  _Float16* Kh  = Qh  + QPAD * 128;       // [6336][128]
  _Float16* Vt  = Kh  + NKEY * 128;       // [128][6336]
  float*    ao  = (float*)(Vt + 128 * NKEY);  // [2512][128] fp32

  cvt_f16_kernel<<<64, 256, 0, stream>>>(wq, wqh, 16384);
  cvt_f16_kernel<<<64, 256, 0, stream>>>(wk, wkh, 16384);
  cvt_f16_kernel<<<64, 256, 0, stream>>>(wv, wvh, 16384);
  cvt_f16_kernel<<<64, 256, 0, stream>>>(wproj, wph, 16384);
  cvt_f16_kernel<<<128, 256, 0, stream>>>(w1, w1h, 32768);
  cvt_f16_kernel<<<128, 256, 0, stream>>>(w2, w2h, 32768);

  ln_proj_kernel<<<QTILES, 32, 0, stream>>>(q, 0, qn_g, qn_b, wqh, bq, Qh, nullptr);
  ln_proj_kernel<<<KTILES, 32, 0, stream>>>(k, 1, kn_g, kn_b, wkh, bk, Kh, nullptr);
  ln_proj_kernel<<<KTILES, 32, 0, stream>>>(v, 1, vn_g, vn_b, wvh, bv, nullptr, Vt);

  attn_kernel<<<dim3(QTILES, NHEADS), 32, 0, stream>>>(Qh, Kh, Vt, Wl, vis, ao);

  post_kernel<<<QTILES, 32, 0, stream>>>(ao, skip, wph, bproj, pre_g, pre_b,
                                         w1h, b1, w2h, b2, post_g, post_b,
                                         (float*)d_out);
}
